// MambaBlockSequential_61744449848168
// MI455X (gfx1250) — matrix operand order repaired and checked by reading, not compiled, not run on hardware
//
#include <hip/hip_runtime.h>

// ---------------------------------------------------------------------------
// Mamba block for MI455X (gfx1250): bf16 WMMA GEMMs + chunked SSM scan.
// Fixed sizes from the reference: B=2, L=2048, D=1024, N=16.
// ---------------------------------------------------------------------------

typedef __attribute__((ext_vector_type(16))) __bf16 v16bf;
typedef __attribute__((ext_vector_type(8)))  float  v8f;

constexpr int BB  = 2;
constexpr int LL  = 2048;
constexpr int DD  = 1024;
constexpr int NN  = 16;
constexpr int MM  = BB * LL;      // 4096 rows in every GEMM
constexpr int NCH = 16;           // scan chunks
constexpr int CHL = LL / NCH;     // 128 steps per chunk

enum { ACT_NONE = 0, ACT_SIGMOID = 1, ACT_SOFTPLUS = 2 };

// pack two floats into one dword of two bf16 (round-to-nearest-even),
// lo -> [15:0], hi -> [31:16]
__device__ __forceinline__ unsigned pack2bf(float lo, float hi) {
  unsigned ul = __builtin_bit_cast(unsigned, lo);
  unsigned uh = __builtin_bit_cast(unsigned, hi);
  ul += 0x7FFFu + ((ul >> 16) & 1u);
  uh += 0x7FFFu + ((uh >> 16) & 1u);
  return (ul >> 16) | (uh & 0xFFFF0000u);
}

// ---------------------------------------------------------------------------
// Tiled GEMM: Y[M,N] = act(X[M,K] @ W[K,N] + bias[N])
// Block = 128 threads = 4 waves. Block tile: 64 rows x (16*NT) cols.
// Each wave owns a 16-row strip and NT accumulators; K steps by 32 (bf16 WMMA).
// A/B tiles staged in double-buffered LDS in the exact WMMA fragment layout:
//   - A: 8 consecutive K -> 8 consecutive fragment elems => one ds_store_b128
//   - B: 2 consecutive K -> one fragment dword           => one ds_store_b32
// One barrier per K-step; each wave reads fragments as aligned v16bf
// (2x ds_load_b128 per fragment).
// ---------------------------------------------------------------------------
template <int ACT, int NT>
__global__ __launch_bounds__(128)
void wmma_gemm(const float* __restrict__ X, const float* __restrict__ W,
               const float* __restrict__ bias, float* __restrict__ Y,
               int M, int K, int N)
{
  __shared__ __align__(32) __bf16 la[2][4][32][16];   // A tile: 64 rows x 32 k
  __shared__ __align__(32) __bf16 lb[2][NT][32][16];  // B tile: 32 k x 16*NT n

  const int tid  = threadIdx.x;
  const int wid  = tid >> 5;
  const int lane = tid & 31;
  const int ln   = lane & 15;
  const int lh   = lane >> 4;

  const int rowBlk = blockIdx.y * 64;
  const int colBlk = blockIdx.x * (16 * NT);
  const int rowW   = rowBlk + wid * 16;

  v8f acc[NT] = {};

  for (int kt = 0, it = 0; kt < K; kt += 32, ++it) {
    const int pb = it & 1;

    // --- A tile: 256 octets (8 consecutive k) -> one b128 LDS store each ---
#pragma unroll
    for (int j = 0; j < 2; ++j) {
      int o = j * 128 + tid;                    // octet id 0..255
      int r = o >> 2;                           // local row 0..63
      int q = o & 3;                            // k-octet 0..3
      const float4* src = (const float4*)(&X[(rowBlk + r) * K + kt + q * 8]);
      float4 f0 = src[0];
      float4 f1 = src[1];
      uint4 pk;
      pk.x = pack2bf(f0.x, f0.y);
      pk.y = pack2bf(f0.z, f0.w);
      pk.z = pack2bf(f1.x, f1.y);
      pk.w = pack2bf(f1.z, f1.w);
      // frag mapping: q&1 -> lane half, q>>1 -> element octet
      *(uint4*)(&la[pb][r >> 4][(q & 1) * 16 + (r & 15)][(q >> 1) * 8]) = pk;
    }

    // --- B tile: k-pairs -> one b32 LDS store each (coalesced in n) ---
#pragma unroll
    for (int j = 0; j < 2 * NT; ++j) {
      int pid = j * 128 + tid;                  // 0..256*NT-1
      int kk  = (pid / (16 * NT)) * 2;          // even k 0..30
      int n   = pid % (16 * NT);
      float w0 = W[(kt + kk)     * N + colBlk + n];
      float w1 = W[(kt + kk + 1) * N + colBlk + n];
      *(unsigned*)(&lb[pb][n >> 4][((kk >> 3) & 1) * 16 + (n & 15)][(kk & 7) + (kk >> 4) * 8]) =
          pack2bf(w0, w1);
    }
    __syncthreads();   // single barrier per step (double-buffered tiles)

    v16bf afrag = *(const v16bf*)(&la[pb][wid][lane][0]);
#pragma unroll
    for (int t = 0; t < NT; ++t) {
      v16bf bfrag = *(const v16bf*)(&lb[pb][t][lane][0]);
      acc[t] = __builtin_amdgcn_wmma_f32_16x16x32_bf16(
          false, afrag, false, bfrag, (short)0, acc[t], false, false);
    }

    if (kt + 32 < K) {   // hint next activation tile (global_prefetch_b8)
      __builtin_prefetch(&X[(rowBlk + (tid >> 1)) * K + kt + 32], 0, 1);
    }
  }

  // --- epilogue: bias + activation, D-matrix layout (m = r + 8*lane_half) ---
#pragma unroll
  for (int t = 0; t < NT; ++t) {
    int   n  = colBlk + t * 16 + ln;
    float bv = bias[n];
#pragma unroll
    for (int r = 0; r < 8; ++r) {
      int   m = rowW + r + 8 * lh;
      float v = acc[t][r] + bv;
      if (ACT == ACT_SIGMOID)       v = 1.0f / (1.0f + __expf(-v));
      else if (ACT == ACT_SOFTPLUS) v = (v > 20.0f) ? v : log1pf(__expf(v));
      Y[m * N + n] = v;
    }
  }
}

// ---------------------------------------------------------------------------
// Chunked SSM scan. Recurrence: h = exp(dt*A)*h + (x_ssm*dt)*B;  y = <h, C>.
// Chunk decay product collapses: prod_l exp(dt_l*A[n]) = exp(A[n]*sum(dt)).
// ---------------------------------------------------------------------------

// Phase 1: per chunk, local state (h0=0) and scalar dt-sum.
__global__ __launch_bounds__(256)
void scan_phase1(const float* __restrict__ xs, const float* __restrict__ dtb,
                 const float* __restrict__ Bm, const float* __restrict__ A_log,
                 float* __restrict__ hC, float* __restrict__ Sdt)
{
  const int tid = blockIdx.x * blockDim.x + threadIdx.x;  // 0..NCH*BB*DD-1
  const int c   = tid / (BB * DD);
  const int r   = tid % (BB * DD);
  const int b   = r / DD;
  const int d   = r % DD;

  float A[NN];
#pragma unroll
  for (int n = 0; n < NN; ++n) A[n] = -__expf(A_log[d * NN + n]);

  float h[NN] = {};
  float s = 0.0f;
  int base = (b * LL + c * CHL) * DD + d;
  int bc   = (b * LL + c * CHL) * NN;
  for (int l = 0; l < CHL; ++l) {
    float dtv = dtb[base];
    float u   = xs[base] * dtv;
    s += dtv;
    float bv[NN];
#pragma unroll
    for (int n = 0; n < NN; ++n) bv[n] = Bm[bc + n];
#pragma unroll
    for (int n = 0; n < NN; ++n) {
      float a = __expf(dtv * A[n]);
      h[n] = fmaf(a, h[n], u * bv[n]);
    }
    base += DD; bc += NN;
  }
  const int off = (c * BB * DD + r) * NN;
#pragma unroll
  for (int n = 0; n < NN; ++n) hC[off + n] = h[n];
  Sdt[c * BB * DD + r] = s;
}

// Phase 2 (serial over 16 chunks only): carry-in state per chunk.
__global__ __launch_bounds__(256)
void scan_combine(const float* __restrict__ hC, const float* __restrict__ Sdt,
                  const float* __restrict__ A_log, float* __restrict__ hIn)
{
  const int t = blockIdx.x * blockDim.x + threadIdx.x;  // 0..BB*DD-1
  const int d = t % DD;

  float A[NN];
#pragma unroll
  for (int n = 0; n < NN; ++n) A[n] = -__expf(A_log[d * NN + n]);

  float h[NN] = {};
  for (int c = 0; c < NCH; ++c) {
    const int off = (c * BB * DD + t) * NN;
#pragma unroll
    for (int n = 0; n < NN; ++n) hIn[off + n] = h[n];
    float s = Sdt[c * BB * DD + t];
#pragma unroll
    for (int n = 0; n < NN; ++n)
      h[n] = fmaf(__expf(A[n] * s), h[n], hC[off + n]);
  }
}

// Phase 3: replay chunks with correct carry-in; emit gated output g = y * z.
__global__ __launch_bounds__(256)
void scan_phase2(const float* __restrict__ xs, const float* __restrict__ dtb,
                 const float* __restrict__ zb, const float* __restrict__ Bm,
                 const float* __restrict__ Cm, const float* __restrict__ A_log,
                 const float* __restrict__ hIn, float* __restrict__ g)
{
  const int tid = blockIdx.x * blockDim.x + threadIdx.x;
  const int c   = tid / (BB * DD);
  const int r   = tid % (BB * DD);
  const int b   = r / DD;
  const int d   = r % DD;

  float A[NN];
#pragma unroll
  for (int n = 0; n < NN; ++n) A[n] = -__expf(A_log[d * NN + n]);

  float h[NN];
  const int off = (c * BB * DD + r) * NN;
#pragma unroll
  for (int n = 0; n < NN; ++n) h[n] = hIn[off + n];

  int base = (b * LL + c * CHL) * DD + d;
  int bc   = (b * LL + c * CHL) * NN;
  for (int l = 0; l < CHL; ++l) {
    float dtv = dtb[base];
    float u   = xs[base] * dtv;
    float zv  = zb[base];
    float bv[NN], cv[NN];
#pragma unroll
    for (int n = 0; n < NN; ++n) { bv[n] = Bm[bc + n]; cv[n] = Cm[bc + n]; }
    float y = 0.0f;
#pragma unroll
    for (int n = 0; n < NN; ++n) {
      float a = __expf(dtv * A[n]);
      h[n] = fmaf(a, h[n], u * bv[n]);
      y    = fmaf(h[n], cv[n], y);
    }
    g[base] = y * zv;
    base += DD; bc += NN;
  }
}

// ---------------------------------------------------------------------------
// Launcher
// ---------------------------------------------------------------------------
extern "C" void kernel_launch(void* const* d_in, const int* in_sizes, int n_in,
                              void* d_out, int out_size, void* d_ws, size_t ws_size,
                              hipStream_t stream) {
  (void)in_sizes; (void)n_in; (void)out_size; (void)ws_size;

  const float* x     = (const float*)d_in[0];
  const float* Wz    = (const float*)d_in[1];
  const float* bz    = (const float*)d_in[2];
  const float* Wx    = (const float*)d_in[3];
  const float* bx    = (const float*)d_in[4];
  const float* WB    = (const float*)d_in[5];
  const float* bB    = (const float*)d_in[6];
  const float* WC    = (const float*)d_in[7];
  const float* bC    = (const float*)d_in[8];
  const float* Wdt   = (const float*)d_in[9];
  const float* bdt   = (const float*)d_in[10];
  const float* A_log = (const float*)d_in[11];
  const float* Wout  = (const float*)d_in[12];
  const float* bout  = (const float*)d_in[13];
  float* out = (float*)d_out;

  // workspace partition (~72 MB of fp32)
  float* zbuf  = (float*)d_ws;                       // MM*DD
  float* xsbuf = zbuf  + (size_t)MM * DD;            // MM*DD
  float* dtbuf = xsbuf + (size_t)MM * DD;            // MM*DD
  float* gbuf  = dtbuf + (size_t)MM * DD;            // MM*DD
  float* Bmbuf = gbuf  + (size_t)MM * DD;            // MM*NN
  float* Cbuf  = Bmbuf + (size_t)MM * NN;            // MM*NN
  float* hC    = Cbuf  + (size_t)MM * NN;            // NCH*BB*DD*NN
  float* hIn   = hC    + (size_t)NCH * BB * DD * NN; // NCH*BB*DD*NN
  float* Sdt   = hIn   + (size_t)NCH * BB * DD * NN; // NCH*BB*DD

  dim3 blk(128);
  dim3 gBig(DD / 64, MM / 64);   // (16, 64)
  dim3 gSmall(1, MM / 64);       // N=16 -> one col tile

  // input-dependent projections (fused bias + activation)
  wmma_gemm<ACT_SIGMOID , 4><<<gBig,   blk, 0, stream>>>(x, Wz , bz , zbuf , MM, DD, DD);
  wmma_gemm<ACT_NONE    , 4><<<gBig,   blk, 0, stream>>>(x, Wx , bx , xsbuf, MM, DD, DD);
  wmma_gemm<ACT_SOFTPLUS, 4><<<gBig,   blk, 0, stream>>>(x, Wdt, bdt, dtbuf, MM, DD, DD);
  wmma_gemm<ACT_NONE    , 1><<<gSmall, blk, 0, stream>>>(x, WB , bB , Bmbuf, MM, DD, NN);
  wmma_gemm<ACT_NONE    , 1><<<gSmall, blk, 0, stream>>>(x, WC , bC , Cbuf , MM, DD, NN);

  // chunked selective scan (16x parallelism over L)
  scan_phase1 <<<(NCH * BB * DD) / 256, 256, 0, stream>>>(xsbuf, dtbuf, Bmbuf, A_log, hC, Sdt);
  scan_combine<<<(BB * DD) / 256,        256, 0, stream>>>(hC, Sdt, A_log, hIn);
  scan_phase2 <<<(NCH * BB * DD) / 256, 256, 0, stream>>>(xsbuf, dtbuf, zbuf, Bmbuf, Cbuf,
                                                          A_log, hIn, gbuf);

  // output projection
  wmma_gemm<ACT_NONE, 4><<<gBig, blk, 0, stream>>>(gbuf, Wout, bout, out, MM, DD, DD);
}
